// MTCNN_49168785605411
// MI455X (gfx1250) — matrix operand order repaired and checked by reading, not compiled, not run on hardware
//
#include <hip/hip_runtime.h>
#include <hip/hip_bf16.h>
#include <math.h>

// ---------------------------------------------------------------------------
// CDNA5 WMMA types (wave32): v_wmma_f32_16x16x32_f16
// ---------------------------------------------------------------------------
typedef __attribute__((ext_vector_type(16))) _Float16 v16h;
typedef __attribute__((ext_vector_type(8)))  float    v8f;

// Fragment layouts per CDNA5 ISA 7.12.2 (16-bit, wave32):
//   A 16x32 : lane L -> row M=L%16; elem e (VGPR v=e/2):
//             K = (v>=4?16:0) + (v&3)*2 + (L/16)*8 + (e&1)
//   B 32x16 : lane L -> col N=L%16; elem e: K = (e>=8?16:0) + (L/16)*8 + (e&7)
//   D 16x16 : elem v -> M = v + 8*(L/16), N = L%16
//
// Packed operand memory layout (both A and B):
//   halfbuf[ ((tile*nk + kt)*32 + lane)*16 + e ]   (32 bytes per lane per chunk)

// ---------------------------------------------------------------------------
// Pack conv/dense weights (OIHW f32) into A-fragment f16 layout. Done once.
// ---------------------------------------------------------------------------
__global__ void pack_a_kernel(const float* __restrict__ w, _Float16* __restrict__ out,
                              int OC, int C, int KH, int KW, int mtiles, int nk)
{
    const long idx = (long)blockIdx.x * blockDim.x + threadIdx.x;
    const long total = (long)mtiles * nk * 512;
    if (idx >= total) return;
    const int e    = (int)(idx & 15);
    const int lane = (int)((idx >> 4) & 31);
    const long t   = idx >> 9;                 // mt*nk + kt
    const int kt   = (int)(t % nk);
    const int mt   = (int)(t / nk);
    const int half = lane >> 4, lq = lane & 15;
    const int KK = KH * KW, Ktot = C * KK;
    const int v  = e >> 1;
    const int ka = kt * 32 + ((v >= 4) ? 16 : 0) + ((v & 3) << 1) + (half << 3) + (e & 1);
    const int m  = mt * 16 + lq;
    float val = 0.f;
    if (m < OC && ka < Ktot) {
        int ic = ka / KK; int rr = ka - ic * KK;
        int kh = rr / KW; int kw = rr - kh * KW;
        val = w[((m * C + ic) * KH + kh) * KW + kw];
    }
    out[idx] = (_Float16)val;
}

// ---------------------------------------------------------------------------
// im2col-pack activations (NCHW f32) into B-fragment f16 layout.
// Absorbs the (x - 127.5)/128 standardization. All the / and % live here,
// outside the GEMM hot loop.
// ---------------------------------------------------------------------------
__global__ void pack_b_kernel(const float* __restrict__ x, _Float16* __restrict__ out,
                              int N, int C, int H, int W, int KH, int KW,
                              int OH, int OW, int nt, int nk, int do_std)
{
    const long idx = (long)blockIdx.x * blockDim.x + threadIdx.x;
    const long total = (long)nt * nk * 512;
    if (idx >= total) return;
    const int e    = (int)(idx & 15);
    const int lane = (int)((idx >> 4) & 31);
    const long t   = idx >> 9;                 // pt*nk + kt
    const int kt   = (int)(t % nk);
    const long pt  = t / nk;
    const int half = lane >> 4, lq = lane & 15;
    const int S = OH * OW;
    const long P = (long)N * S;
    const long p = pt * 16 + lq;
    const int kb = kt * 32 + ((e >= 8) ? 16 : 0) + (half << 3) + (e & 7);
    const int KK = KH * KW, Ktot = C * KK;
    float val = 0.f;
    if (p < P && kb < Ktot) {
        int n = (int)(p / S); int r = (int)(p - (long)n * S);
        int oh = r / OW, ow = r - (r / OW) * OW;
        int ic = kb / KK; int rr = kb - ic * KK;
        int kh = rr / KW; int kw = rr - kh * KW;
        val = x[(((long)n * C + ic) * H + (oh + kh)) * W + (ow + kw)];
        if (do_std) val = (val - 127.5f) * 0.0078125f;
    }
    out[idx] = (_Float16)val;
}

// ---------------------------------------------------------------------------
// Dense GEMM on packed operands. 4 waves/block; each wave: 16(M) x 64(N)
// tile = 4 accumulators, one shared A fragment per K-chunk -> 4 v_wmma per
// iteration, all operand loads are 32B vector loads.
// Epilogue: bias + optional PReLU, scatter to NCHW output.
// ---------------------------------------------------------------------------
__global__ void __launch_bounds__(128)
gemm_wmma_kernel(const _Float16* __restrict__ Apack, const _Float16* __restrict__ Bpack,
                 const float* __restrict__ bias, const float* __restrict__ pre,
                 float* __restrict__ y,
                 int nk, int nt, int P, int S, int OC, int OH, int OW)
{
    const int lane = threadIdx.x & 31;
    const int wv   = (threadIdx.x >> 5) & 3;
    const int half = lane >> 4;
    const int lq   = lane & 15;
    const int mt   = blockIdx.y;

    const int nt0 = blockIdx.x * 16 + wv * 4;
    const int last = nt - 1;
    const int t0 = nt0, t1 = nt0 + 1, t2 = nt0 + 2, t3 = nt0 + 3;
    const int c0 = t0 > last ? last : t0;   // clamped tiles for uniform WMMA
    const int c1 = t1 > last ? last : t1;
    const int c2 = t2 > last ? last : t2;
    const int c3 = t3 > last ? last : t3;

    const v16h* A = (const v16h*)Apack;
    const v16h* B = (const v16h*)Bpack;

    v8f acc0 = {0.f,0.f,0.f,0.f,0.f,0.f,0.f,0.f};
    v8f acc1 = acc0, acc2 = acc0, acc3 = acc0;

    for (int kt = 0; kt < nk; ++kt) {
        const v16h a  = A[((long)mt * nk + kt) * 32 + lane];
        const v16h b0 = B[((long)c0 * nk + kt) * 32 + lane];
        const v16h b1 = B[((long)c1 * nk + kt) * 32 + lane];
        const v16h b2 = B[((long)c2 * nk + kt) * 32 + lane];
        const v16h b3 = B[((long)c3 * nk + kt) * 32 + lane];
        if (kt + 1 < nk) {  // stream the next K-chunk toward L2/L0
            __builtin_prefetch((const void*)&B[((long)c0 * nk + kt + 1) * 32 + lane], 0, 1);
        }
        acc0 = __builtin_amdgcn_wmma_f32_16x16x32_f16(false, a, false, b0, (short)0, acc0, false, false);
        acc1 = __builtin_amdgcn_wmma_f32_16x16x32_f16(false, a, false, b1, (short)0, acc1, false, false);
        acc2 = __builtin_amdgcn_wmma_f32_16x16x32_f16(false, a, false, b2, (short)0, acc2, false, false);
        acc3 = __builtin_amdgcn_wmma_f32_16x16x32_f16(false, a, false, b3, (short)0, acc3, false, false);
    }

    auto store_tile = [&](const v8f& acc, int tile) {
        if (tile >= nt) return;
        const int p = tile * 16 + lq;
        if (p >= P) return;
        const int n = p / S; const int r = p - n * S;
        const int oh = r / OW, ow = r - (r / OW) * OW;
#pragma unroll
        for (int v = 0; v < 8; ++v) {
            const int m = mt * 16 + v + (half << 3);
            if (m < OC) {
                float o = acc[v] + bias[m];
                if (pre) { float al = pre[m]; o = (o > 0.f) ? o : o * al; }
                y[(((long)n * OC + m) * OH + oh) * OW + ow] = o;
            }
        }
    };
    store_tile(acc0, t0);
    store_tile(acc1, t1);
    store_tile(acc2, t2);
    store_tile(acc3, t3);
}

// ---------------------------------------------------------------------------
// Max-pool, torch ceil_mode=True (right/bottom -inf padding).
// ---------------------------------------------------------------------------
__global__ void maxpool_kernel(const float* __restrict__ x, float* __restrict__ y,
                               int N, int C, int H, int W, int k, int s,
                               int OH, int OW)
{
    const int idx = blockIdx.x * blockDim.x + threadIdx.x;
    const int total = N * C * OH * OW;
    if (idx >= total) return;
    int ow = idx % OW; int t = idx / OW;
    int oh = t % OH;   t /= OH;
    int c  = t % C;    int n = t / C;
    const float* xp = x + ((size_t)(n * C + c)) * H * W;
    const int h0 = oh * s, w0 = ow * s;
    float m = -3.402823466e+38f;
    for (int i = 0; i < k; ++i) {
        int h = h0 + i; if (h >= H) break;
        for (int j = 0; j < k; ++j) {
            int ww = w0 + j; if (ww >= W) break;
            float v = xp[h * W + ww];
            m = (v > m) ? v : m;
        }
    }
    y[idx] = m;
}

// Softmax over 2 channels of NCHW (PNet head). Layout (N, 2, S).
__global__ void softmax_c2_kernel(const float* __restrict__ in, float* __restrict__ out,
                                  int N, int S)
{
    const int idx = blockIdx.x * blockDim.x + threadIdx.x;
    if (idx >= N * S) return;
    const int n = idx / S, sp = idx - n * S;
    const size_t base = (size_t)n * 2 * S;
    float z0 = in[base + sp], z1 = in[base + S + sp];
    float mx = z0 > z1 ? z0 : z1;
    float e0 = __expf(z0 - mx), e1 = __expf(z1 - mx);
    float inv = 1.f / (e0 + e1);
    out[base + sp]     = e0 * inv;
    out[base + S + sp] = e1 * inv;
}

// Softmax over 2 features per row (RNet/ONet heads).
__global__ void softmax_f2_kernel(const float* __restrict__ in, float* __restrict__ out,
                                  int rows)
{
    const int i = blockIdx.x * blockDim.x + threadIdx.x;
    if (i >= rows) return;
    float z0 = in[i * 2], z1 = in[i * 2 + 1];
    float mx = z0 > z1 ? z0 : z1;
    float e0 = __expf(z0 - mx), e1 = __expf(z1 - mx);
    float inv = 1.f / (e0 + e1);
    out[i * 2]     = e0 * inv;
    out[i * 2 + 1] = e1 * inv;
}

// torch x.permute(0,3,2,1).reshape(N,-1): NCHW -> (N, W*H*C) flat.
__global__ void flatten_perm_kernel(const float* __restrict__ in, float* __restrict__ out,
                                    int N, int C, int H, int W)
{
    const int idx = blockIdx.x * blockDim.x + threadIdx.x;
    const int F = C * H * W;
    if (idx >= N * F) return;
    const int n = idx / F, f = idx - n * F;
    const int wi = f / (H * C);
    const int rem = f - wi * (H * C);
    const int hi = rem / C;
    const int ci = rem - hi * C;
    out[idx] = in[(((size_t)n * C + ci) * H + hi) * W + wi];
}

// ---------------------------------------------------------------------------
// Host-side helpers
// ---------------------------------------------------------------------------
struct ConvPack { _Float16* apack; int mt; int nk; };

static ConvPack make_pack(hipStream_t st, _Float16*& cur, const float* w,
                          int OC, int C, int K)
{
    ConvPack cp;
    cp.mt = (OC + 15) / 16;
    cp.nk = (C * K * K + 31) / 32;
    cp.apack = cur;
    const long total = (long)cp.mt * cp.nk * 512;
    pack_a_kernel<<<dim3((unsigned)((total + 255) / 256)), dim3(256), 0, st>>>(
        w, cp.apack, OC, C, K, K, cp.mt, cp.nk);
    cur += total;
    return cp;
}

static void launch_conv(hipStream_t st, const float* x, const ConvPack& wp,
                        const float* b, const float* pre, _Float16* bpack, float* y,
                        int N, int C, int H, int W, int OC, int K, int do_std)
{
    const int OH = H - K + 1, OW = W - K + 1;
    const long P = (long)N * OH * OW;
    const int nt = (int)((P + 15) / 16);
    const long totB = (long)nt * wp.nk * 512;
    pack_b_kernel<<<dim3((unsigned)((totB + 255) / 256)), dim3(256), 0, st>>>(
        x, bpack, N, C, H, W, K, K, OH, OW, nt, wp.nk, do_std);
    dim3 grid((unsigned)((nt + 15) / 16), (unsigned)wp.mt, 1);
    gemm_wmma_kernel<<<grid, dim3(128, 1, 1), 0, st>>>(
        wp.apack, bpack, b, pre, y, wp.nk, nt, (int)P, OH * OW, OC, OH, OW);
}

static void launch_dense(hipStream_t st, const float* x, const ConvPack& wp,
                         const float* b, const float* pre, _Float16* bpack, float* y,
                         int N, int F, int OC)
{
    launch_conv(st, x, wp, b, pre, bpack, y, N, F, 1, 1, OC, 1, 0);
}

static int pool_out(int n, int k, int s) { return (n - k + s - 1) / s + 1; } // ceil mode

static void launch_pool(hipStream_t st, const float* x, float* y,
                        int N, int C, int H, int W, int k, int s)
{
    const int OH = pool_out(H, k, s), OW = pool_out(W, k, s);
    const long total = (long)N * C * OH * OW;
    maxpool_kernel<<<dim3((unsigned)((total + 255) / 256)), dim3(256), 0, st>>>(
        x, y, N, C, H, W, k, s, OH, OW);
}

// ---------------------------------------------------------------------------
extern "C" void kernel_launch(void* const* d_in, const int* in_sizes, int n_in,
                              void* d_out, int out_size, void* d_ws, size_t ws_size,
                              hipStream_t stream)
{
    (void)in_sizes; (void)n_in; (void)out_size; (void)ws_size;

    int ii = 0;
    const float* img = (const float*)d_in[ii++];   // (16,3,512,512)
    const float* c24 = (const float*)d_in[ii++];   // (4096,3,24,24)
    const float* c48 = (const float*)d_in[ii++];   // (2048,3,48,48)

    // PNet params
    const float *Pc1w=(const float*)d_in[ii++], *Pc1b=(const float*)d_in[ii++], *Pp1=(const float*)d_in[ii++];
    const float *Pc2w=(const float*)d_in[ii++], *Pc2b=(const float*)d_in[ii++], *Pp2=(const float*)d_in[ii++];
    const float *Pc3w=(const float*)d_in[ii++], *Pc3b=(const float*)d_in[ii++], *Pp3=(const float*)d_in[ii++];
    const float *Pc41w=(const float*)d_in[ii++], *Pc41b=(const float*)d_in[ii++];
    const float *Pc42w=(const float*)d_in[ii++], *Pc42b=(const float*)d_in[ii++];
    // RNet params
    const float *Rc1w=(const float*)d_in[ii++], *Rc1b=(const float*)d_in[ii++], *Rp1=(const float*)d_in[ii++];
    const float *Rc2w=(const float*)d_in[ii++], *Rc2b=(const float*)d_in[ii++], *Rp2=(const float*)d_in[ii++];
    const float *Rc3w=(const float*)d_in[ii++], *Rc3b=(const float*)d_in[ii++], *Rp3=(const float*)d_in[ii++];
    const float *Rd4w=(const float*)d_in[ii++], *Rd4b=(const float*)d_in[ii++], *Rp4=(const float*)d_in[ii++];
    const float *Rd51w=(const float*)d_in[ii++], *Rd51b=(const float*)d_in[ii++];
    const float *Rd52w=(const float*)d_in[ii++], *Rd52b=(const float*)d_in[ii++];
    // ONet params
    const float *Oc1w=(const float*)d_in[ii++], *Oc1b=(const float*)d_in[ii++], *Op1=(const float*)d_in[ii++];
    const float *Oc2w=(const float*)d_in[ii++], *Oc2b=(const float*)d_in[ii++], *Op2=(const float*)d_in[ii++];
    const float *Oc3w=(const float*)d_in[ii++], *Oc3b=(const float*)d_in[ii++], *Op3=(const float*)d_in[ii++];
    const float *Oc4w=(const float*)d_in[ii++], *Oc4b=(const float*)d_in[ii++], *Op4=(const float*)d_in[ii++];
    const float *Od5w=(const float*)d_in[ii++], *Od5b=(const float*)d_in[ii++], *Op5=(const float*)d_in[ii++];
    const float *Od61w=(const float*)d_in[ii++], *Od61b=(const float*)d_in[ii++];
    const float *Od62w=(const float*)d_in[ii++], *Od62b=(const float*)d_in[ii++];
    const float *Od63w=(const float*)d_in[ii++], *Od63b=(const float*)d_in[ii++];

    float* outf = (float*)d_out;

    // Workspace layout:
    //  [ weight packs : 2^21 halves ][ B pack arena : 2^24 halves ][ stage f32 arena ]
    const long WPACK_HALVES = 1L << 21;      // 4 MB, need ~0.52M halves
    const long BPACK_HALVES = 1L << 24;      // 32 MB, max stage needs ~16.3M halves
    _Float16* wpack_cur = (_Float16*)d_ws;
    _Float16* bpack     = (_Float16*)d_ws + WPACK_HALVES;
    float*    wsf       = (float*)((char*)d_ws + (size_t)(WPACK_HALVES + BPACK_HALVES) * 2);

    // ---- pack all weights once (constant across chunks) ----
    ConvPack PKc1  = make_pack(stream, wpack_cur, Pc1w, 10, 3, 3);
    ConvPack PKc2  = make_pack(stream, wpack_cur, Pc2w, 16, 10, 3);
    ConvPack PKc3  = make_pack(stream, wpack_cur, Pc3w, 32, 16, 3);
    ConvPack PKc41 = make_pack(stream, wpack_cur, Pc41w, 2, 32, 1);
    ConvPack PKc42 = make_pack(stream, wpack_cur, Pc42w, 4, 32, 1);
    ConvPack RKc1  = make_pack(stream, wpack_cur, Rc1w, 28, 3, 3);
    ConvPack RKc2  = make_pack(stream, wpack_cur, Rc2w, 48, 28, 3);
    ConvPack RKc3  = make_pack(stream, wpack_cur, Rc3w, 64, 48, 2);
    ConvPack RKd4  = make_pack(stream, wpack_cur, Rd4w, 128, 576, 1);
    ConvPack RKd51 = make_pack(stream, wpack_cur, Rd51w, 2, 128, 1);
    ConvPack RKd52 = make_pack(stream, wpack_cur, Rd52w, 4, 128, 1);
    ConvPack OKc1  = make_pack(stream, wpack_cur, Oc1w, 32, 3, 3);
    ConvPack OKc2  = make_pack(stream, wpack_cur, Oc2w, 64, 32, 3);
    ConvPack OKc3  = make_pack(stream, wpack_cur, Oc3w, 64, 64, 3);
    ConvPack OKc4  = make_pack(stream, wpack_cur, Oc4w, 128, 64, 2);
    ConvPack OKd5  = make_pack(stream, wpack_cur, Od5w, 256, 1152, 1);
    ConvPack OKd61 = make_pack(stream, wpack_cur, Od61w, 2, 256, 1);
    ConvPack OKd62 = make_pack(stream, wpack_cur, Od62w, 4, 256, 1);
    ConvPack OKd63 = make_pack(stream, wpack_cur, Od63w, 10, 256, 1);

    // Output regions (return order: pb, pa, rb, ra, ob, oc, oa)
    const size_t PB_OFF = 0;
    const size_t PA_OFF = PB_OFF + (size_t)16 * 4 * 251 * 251;
    const size_t RB_OFF = PA_OFF + (size_t)16 * 2 * 251 * 251;
    const size_t RA_OFF = RB_OFF + (size_t)4096 * 4;
    const size_t OB_OFF = RA_OFF + (size_t)4096 * 2;
    const size_t OC_OFF = OB_OFF + (size_t)2048 * 4;
    const size_t OA_OFF = OC_OFF + (size_t)2048 * 10;

    // ================= PNet: 16 images, chunk = 1 image =================
    {
        float* t1 = wsf;                               // 10*510*510
        float* t2 = t1 + (size_t)10 * 510 * 510;       // 10*255*255
        float* t3 = t2 + (size_t)10 * 255 * 255;       // 16*253*253
        float* t4 = t3 + (size_t)16 * 253 * 253;       // 32*251*251
        float* t5 = t4 + (size_t)32 * 251 * 251;       // 2*251*251
        for (int n = 0; n < 16; ++n) {
            const float* xin = img + (size_t)n * 3 * 512 * 512;
            launch_conv(stream, xin, PKc1, Pc1b, Pp1, bpack, t1, 1, 3, 512, 512, 10, 3, 1);
            launch_pool(stream, t1, t2, 1, 10, 510, 510, 2, 2);                 // ->255
            launch_conv(stream, t2, PKc2, Pc2b, Pp2, bpack, t3, 1, 10, 255, 255, 16, 3, 0);
            launch_conv(stream, t3, PKc3, Pc3b, Pp3, bpack, t4, 1, 16, 253, 253, 32, 3, 0);
            launch_conv(stream, t4, PKc41, Pc41b, nullptr, bpack, t5, 1, 32, 251, 251, 2, 1, 0);
            launch_conv(stream, t4, PKc42, Pc42b, nullptr, bpack,
                        outf + PB_OFF + (size_t)n * 4 * 251 * 251,
                        1, 32, 251, 251, 4, 1, 0);
            const int S = 251 * 251;
            softmax_c2_kernel<<<dim3((S + 255) / 256), dim3(256), 0, stream>>>(
                t5, outf + PA_OFF + (size_t)n * 2 * S, 1, S);
        }
    }

    // ================= RNet: 4096 crops, chunk = 512 =================
    {
        const int CH = 512;
        float* t1 = wsf;                                // 512*28*22*22
        float* t2 = t1 + (size_t)CH * 28 * 22 * 22;     // 512*28*11*11
        float* t3 = t2 + (size_t)CH * 28 * 11 * 11;     // 512*48*9*9
        float* t4 = t3 + (size_t)CH * 48 * 9 * 9;       // 512*48*4*4
        float* t5 = t4 + (size_t)CH * 48 * 4 * 4;       // 512*64*3*3
        float* fl = t5 + (size_t)CH * 64 * 3 * 3;       // 512*576
        float* d4 = fl + (size_t)CH * 576;              // 512*128
        float* s51 = d4 + (size_t)CH * 128;             // 512*2
        for (int c = 0; c < 4096 / CH; ++c) {
            const int s0 = c * CH;
            const float* xin = c24 + (size_t)s0 * 3 * 24 * 24;
            launch_conv(stream, xin, RKc1, Rc1b, Rp1, bpack, t1, CH, 3, 24, 24, 28, 3, 1);  // 22
            launch_pool(stream, t1, t2, CH, 28, 22, 22, 3, 2);                              // 11
            launch_conv(stream, t2, RKc2, Rc2b, Rp2, bpack, t3, CH, 28, 11, 11, 48, 3, 0);  // 9
            launch_pool(stream, t3, t4, CH, 48, 9, 9, 3, 2);                                // 4
            launch_conv(stream, t4, RKc3, Rc3b, Rp3, bpack, t5, CH, 48, 4, 4, 64, 2, 0);    // 3
            {
                const long tot = (long)CH * 64 * 3 * 3;
                flatten_perm_kernel<<<dim3((unsigned)((tot + 255) / 256)), dim3(256), 0, stream>>>(
                    t5, fl, CH, 64, 3, 3);
            }
            launch_dense(stream, fl, RKd4, Rd4b, Rp4, bpack, d4, CH, 576, 128);
            launch_dense(stream, d4, RKd52, Rd52b, nullptr, bpack,
                         outf + RB_OFF + (size_t)s0 * 4, CH, 128, 4);
            launch_dense(stream, d4, RKd51, Rd51b, nullptr, bpack, s51, CH, 128, 2);
            softmax_f2_kernel<<<dim3((CH + 255) / 256), dim3(256), 0, stream>>>(
                s51, outf + RA_OFF + (size_t)s0 * 2, CH);
        }
    }

    // ================= ONet: 2048 crops, chunk = 128 =================
    {
        const int CH = 128;
        float* t1 = wsf;                                 // 128*32*46*46
        float* t2 = t1 + (size_t)CH * 32 * 46 * 46;      // 128*32*23*23
        float* t3 = t2 + (size_t)CH * 32 * 23 * 23;      // 128*64*21*21
        float* t4 = t3 + (size_t)CH * 64 * 21 * 21;      // 128*64*10*10
        float* t5 = t4 + (size_t)CH * 64 * 10 * 10;      // 128*64*8*8
        float* t6 = t5 + (size_t)CH * 64 * 8 * 8;        // 128*64*4*4
        float* t7 = t6 + (size_t)CH * 64 * 4 * 4;        // 128*128*3*3
        float* fl = t7 + (size_t)CH * 128 * 3 * 3;       // 128*1152
        float* d5 = fl + (size_t)CH * 1152;              // 128*256
        float* s61 = d5 + (size_t)CH * 256;              // 128*2
        for (int c = 0; c < 2048 / CH; ++c) {
            const int s0 = c * CH;
            const float* xin = c48 + (size_t)s0 * 3 * 48 * 48;
            launch_conv(stream, xin, OKc1, Oc1b, Op1, bpack, t1, CH, 3, 48, 48, 32, 3, 1);  // 46
            launch_pool(stream, t1, t2, CH, 32, 46, 46, 3, 2);                              // 23
            launch_conv(stream, t2, OKc2, Oc2b, Op2, bpack, t3, CH, 32, 23, 23, 64, 3, 0);  // 21
            launch_pool(stream, t3, t4, CH, 64, 21, 21, 3, 2);                              // 10
            launch_conv(stream, t4, OKc3, Oc3b, Op3, bpack, t5, CH, 64, 10, 10, 64, 3, 0);  // 8
            launch_pool(stream, t5, t6, CH, 64, 8, 8, 2, 2);                                // 4
            launch_conv(stream, t6, OKc4, Oc4b, Op4, bpack, t7, CH, 64, 4, 4, 128, 2, 0);   // 3
            {
                const long tot = (long)CH * 128 * 3 * 3;
                flatten_perm_kernel<<<dim3((unsigned)((tot + 255) / 256)), dim3(256), 0, stream>>>(
                    t7, fl, CH, 128, 3, 3);
            }
            launch_dense(stream, fl, OKd5, Od5b, Op5, bpack, d5, CH, 1152, 256);
            launch_dense(stream, d5, OKd62, Od62b, nullptr, bpack,
                         outf + OB_OFF + (size_t)s0 * 4, CH, 256, 4);
            launch_dense(stream, d5, OKd63, Od63b, nullptr, bpack,
                         outf + OC_OFF + (size_t)s0 * 10, CH, 256, 10);
            launch_dense(stream, d5, OKd61, Od61b, nullptr, bpack, s61, CH, 256, 2);
            softmax_f2_kernel<<<dim3((CH + 255) / 256), dim3(256), 0, stream>>>(
                s61, outf + OA_OFF + (size_t)s0 * 2, CH);
        }
    }
}
